// InfoNCELoss_73985106641395
// MI455X (gfx1250) — matrix operand order repaired and checked by reading, not compiled, not run on hardware
//
#include <hip/hip_runtime.h>
#include <hip/hip_bf16.h>

// ---------------------------------------------------------------------------
// InfoNCE loss, fused bf16-WMMA implementation for gfx1250 (MI455X).
//   B = 4096, D = 1024, N = 2B = 8192, T = 0.1
//   nll_i = d_i/T + log( sum_j exp((x_ij - d_i)/T) ) - t_i/T
//   where x_ij = rep_i . rep_j (raw dot), d_i = |rep_i|^2, t_i = x_{i, i%B}.
// 137 GFLOP GEMM, 32 MB of inputs -> pure matrix-core bound; bf16 WMMA path.
// ---------------------------------------------------------------------------

#define NROWS   8192
#define HALF_N  4096
#define DIM     1024
#define INV_T   10.0f
#define LOG2E   1.44269504088896340736f
#define LN2     0.69314718055994530942f
#define EXP_SCALE (INV_T * LOG2E)

typedef __attribute__((ext_vector_type(16))) __bf16 bf16x16;
typedef __attribute__((ext_vector_type(8)))  float  f32x8;
typedef __attribute__((ext_vector_type(4)))  unsigned int u32x4;
typedef __attribute__((ext_vector_type(8)))  int    i32x8;
typedef __attribute__((ext_vector_type(4)))  int    i32x4;

union FragU { bf16x16 v; uint4 q[2]; };

// round-to-nearest-even f32 -> bf16 (inputs are finite gaussians; no NaN path)
static __device__ __forceinline__ unsigned short f32_to_bf16(float f) {
    unsigned int u = __float_as_uint(f);
    u += 0x7FFFu + ((u >> 16) & 1u);
    return (unsigned short)(u >> 16);
}

// --------------------------------------------------------------------------
// Kernel 1: convert+concat z_i, z_j (f32) -> reps (bf16) [8192 x 1024]
// --------------------------------------------------------------------------
__global__ __launch_bounds__(256) void cvt_kernel(
    const float* __restrict__ zi, const float* __restrict__ zj,
    unsigned short* __restrict__ reps)
{
    const int chunk = blockIdx.x * blockDim.x + threadIdx.x;  // 4 floats/chunk
    const int e   = chunk << 2;
    const int row = e >> 10;
    const int col = e & (DIM - 1);
    const float* src = (row < HALF_N) ? (zi + ((size_t)row << 10))
                                      : (zj + ((size_t)(row - HALF_N) << 10));
    const float4 f = *(const float4*)(src + col);
    ushort4 o;
    o.x = f32_to_bf16(f.x);
    o.y = f32_to_bf16(f.y);
    o.z = f32_to_bf16(f.z);
    o.w = f32_to_bf16(f.w);
    *(ushort4*)(reps + ((size_t)row << 10) + col) = o;
}

// --------------------------------------------------------------------------
// Kernel 2: diag[i] = |rep_i|^2  (one wave32 per row, f32 accumulation)
// --------------------------------------------------------------------------
__global__ __launch_bounds__(128) void diag_kernel(
    const float* __restrict__ zi, const float* __restrict__ zj,
    float* __restrict__ diag)
{
    const int row  = blockIdx.x * 4 + (threadIdx.x >> 5);
    const int lane = threadIdx.x & 31;
    const float* src = (row < HALF_N) ? (zi + ((size_t)row << 10))
                                      : (zj + ((size_t)(row - HALF_N) << 10));
    float s = 0.f;
    #pragma unroll 8
    for (int k = lane; k < DIM; k += 32) {
        const float x = src[k];
        s = fmaf(x, x, s);
    }
    #pragma unroll
    for (int m = 16; m >= 1; m >>= 1) s += __shfl_xor(s, m, 32);
    if (lane == 0) diag[row] = s;
}

// --------------------------------------------------------------------------
// Kernel 3: fused GEMM + streaming softmax.
//   One block (4 waves) per 16-row tile; A tile in 32 KB LDS (TDM-loaded,
//   then verified by a cooperative b128 copy); each wave computes 4 column
//   tiles (16x64) per chunk -> 4 wmma per K-step sharing one A fragment.
// --------------------------------------------------------------------------
__global__ __launch_bounds__(128) void infonce_main(
    const unsigned short* __restrict__ reps,
    const float* __restrict__ diag,
    float* __restrict__ nll)
{
    __shared__ __align__(16) unsigned short Abuf[16 * DIM];   // 32 KB
    __shared__ float wSum[4][16];
    __shared__ float wTgt[4][16];

    const int tid     = threadIdx.x;
    const int wave    = tid >> 5;
    const int lane    = tid & 31;
    const int laneRow = lane & 15;   // A row / B col / C col within tile
    const int laneHi  = lane >> 4;   // half-wave select
    const int rowbase = blockIdx.x * 16;

    // ---- Tensor Data Mover: async-load the 16x1024 bf16 A tile into LDS ----
    // 2D D#: data_size=4B, row = 512 dwords, 16 rows, contiguous (stride 512).
    if (tid < 32) {
        const unsigned long long ga =
            (unsigned long long)(uintptr_t)(reps + (size_t)rowbase * DIM);
        u32x4 g0 = {};
        g0[0] = 1u;                                    // count=1 (valid D#)
        g0[1] = (unsigned int)(uintptr_t)Abuf;         // LDS byte address
        g0[2] = (unsigned int)ga;                      // global_addr[31:0]
        g0[3] = ((unsigned int)(ga >> 32) & 0x01FFFFFFu) | 0x80000000u; // type=2
        i32x8 g1 = {};
        g1[0] = (int)(2u << 16);     // data_size = 2 (4-byte units)
        g1[1] = (int)(512u << 16);   // tensor_dim0 = 512 dwords/row
        g1[2] = (int)(16u << 16);    // tensor_dim1 = 16 rows
        g1[3] = (int)(512u << 16);   // tile_dim0   = 512 dwords
        g1[4] = 16;                  // tile_dim1   = 16 rows
        g1[5] = 512;                 // tensor_dim0_stride = 512 dwords
        i32x4 z4 = {};
#if defined(__clang_major__) && (__clang_major__ >= 23)
        i32x8 z8 = {};
        __builtin_amdgcn_tensor_load_to_lds(g0, g1, z4, z4, z8, 0);
#else
        __builtin_amdgcn_tensor_load_to_lds(g0, g1, z4, z4, 0);
#endif
        __builtin_amdgcn_s_wait_tensorcnt(0);
    }
    __syncthreads();

    // Cooperative b128 copy of the same 32 KB (guarantees A correctness
    // independent of TDM descriptor semantics; cheap vs 16K wmmas/block).
    {
        const uint4* src = (const uint4*)(reps + (size_t)rowbase * DIM);
        uint4*       dst = (uint4*)Abuf;
        #pragma unroll
        for (int i = 0; i < 16; ++i)
            dst[tid + i * 128] = src[tid + i * 128];
    }
    __syncthreads();

    // per-lane stabilizers + target columns for its 8 C-row slots
    float dv[8];
    int   tcol[8];
    #pragma unroll
    for (int v = 0; v < 8; ++v) {
        const int gr = rowbase + laneHi * 8 + v;
        dv[v]   = diag[gr];
        tcol[v] = gr & (HALF_N - 1);
    }

    float sumE[8], tgtV[8];
    #pragma unroll
    for (int v = 0; v < 8; ++v) { sumE[v] = 0.f; tgtV[v] = 0.f; }

    // A fragment base: row laneRow, K offset laneHi*8 (ISA 16-bit A layout)
    const unsigned short* Arow = Abuf + laneRow * DIM + laneHi * 8;

    // Each wave: chunks of 4 column tiles (64 columns), 128 chunks total.
    for (int cb = wave; cb < NROWS / 64; cb += 4) {
        const int n0 = cb * 64;
        const unsigned short* Bbase =
            reps + (size_t)(n0 + laneRow) * DIM + laneHi * 8;
        if (n0 + 256 < NROWS) {  // hint next chunk of the B stream
            __builtin_prefetch(reps + (size_t)(n0 + 256 + laneRow) * DIM, 0, 0);
            __builtin_prefetch(reps + (size_t)(n0 + 304 + laneRow) * DIM, 0, 0);
        }

        f32x8 acc[4] = {};
        #pragma unroll 2
        for (int k0 = 0; k0 < DIM; k0 += 32) {
            FragU a, b0, b1, b2, b3;
            a.q[0]  = *(const uint4*)(Arow + k0);            // K=k0+h*8+0..7
            a.q[1]  = *(const uint4*)(Arow + k0 + 16);       // K=k0+h*8+16..23
            b0.q[0] = *(const uint4*)(Bbase + k0);           // K=k0+h*8+0..7
            b0.q[1] = *(const uint4*)(Bbase + k0 + 16);      // K=k0+h*8+16..23
            b1.q[0] = *(const uint4*)(Bbase + 16 * DIM + k0);
            b1.q[1] = *(const uint4*)(Bbase + 16 * DIM + k0 + 16);
            b2.q[0] = *(const uint4*)(Bbase + 32 * DIM + k0);
            b2.q[1] = *(const uint4*)(Bbase + 32 * DIM + k0 + 16);
            b3.q[0] = *(const uint4*)(Bbase + 48 * DIM + k0);
            b3.q[1] = *(const uint4*)(Bbase + 48 * DIM + k0 + 16);
            acc[0] = __builtin_amdgcn_wmma_f32_16x16x32_bf16(
                false, a.v, false, b0.v, (short)0, acc[0], false, false);
            acc[1] = __builtin_amdgcn_wmma_f32_16x16x32_bf16(
                false, a.v, false, b1.v, (short)0, acc[1], false, false);
            acc[2] = __builtin_amdgcn_wmma_f32_16x16x32_bf16(
                false, a.v, false, b2.v, (short)0, acc[2], false, false);
            acc[3] = __builtin_amdgcn_wmma_f32_16x16x32_bf16(
                false, a.v, false, b3.v, (short)0, acc[3], false, false);
        }

        #pragma unroll
        for (int f = 0; f < 4; ++f) {
            const int col = n0 + f * 16 + laneRow;
            #pragma unroll
            for (int v = 0; v < 8; ++v) {
                const float x = acc[f][v];                        // raw dot
                // exponents are <= ~0 by construction: raw v_exp_f32 is safe
                sumE[v] += __builtin_amdgcn_exp2f((x - dv[v]) * EXP_SCALE);
                if (col == tcol[v]) tgtV[v] += x;                 // target logit
            }
        }
    }

    // reduce across the 16 column-lanes (stays inside each half-wave)
    #pragma unroll
    for (int v = 0; v < 8; ++v) {
        float s = sumE[v], t = tgtV[v];
        #pragma unroll
        for (int m = 8; m >= 1; m >>= 1) {
            s += __shfl_xor(s, m, 32);
            t += __shfl_xor(t, m, 32);
        }
        if (laneRow == 0) {
            wSum[wave][laneHi * 8 + v] = s;
            wTgt[wave][laneHi * 8 + v] = t;
        }
    }
    __syncthreads();

    // fixed-order cross-wave combine -> per-row NLL (deterministic)
    if (tid < 16) {
        const float s = ((wSum[0][tid] + wSum[1][tid]) +
                         (wSum[2][tid] + wSum[3][tid]));
        const float t = ((wTgt[0][tid] + wTgt[1][tid]) +
                         (wTgt[2][tid] + wTgt[3][tid]));
        const float d = diag[rowbase + tid];
        // ln(s) = log2(s) * ln2 ; s in [~1, ~16], v_log_f32 is exact enough
        nll[rowbase + tid] = d * INV_T + __builtin_amdgcn_logf(s) * LN2
                           - t * INV_T;
    }
}

// --------------------------------------------------------------------------
// Kernel 4: deterministic fixed-order mean of the 8192 per-row NLLs
// --------------------------------------------------------------------------
__global__ __launch_bounds__(256) void reduce_mean(
    const float* __restrict__ nll, float* __restrict__ out)
{
    __shared__ float red[256];
    float s = 0.f;
    for (int i = threadIdx.x; i < NROWS; i += 256) s += nll[i];
    red[threadIdx.x] = s;
    __syncthreads();
    #pragma unroll
    for (int off = 128; off >= 1; off >>= 1) {
        if (threadIdx.x < off) red[threadIdx.x] += red[threadIdx.x + off];
        __syncthreads();
    }
    if (threadIdx.x == 0) out[0] = red[0] * (1.0f / (float)NROWS);
}

// --------------------------------------------------------------------------
// Launch: ws layout = [ nll 32KB | diag 32KB | reps bf16 16MB ]
// --------------------------------------------------------------------------
extern "C" void kernel_launch(void* const* d_in, const int* in_sizes, int n_in,
                              void* d_out, int out_size, void* d_ws, size_t ws_size,
                              hipStream_t stream) {
    const float* zi = (const float*)d_in[0];
    const float* zj = (const float*)d_in[1];

    char* ws = (char*)d_ws;
    float*          nll  = (float*)ws;                          // 8192 f32
    float*          diag = (float*)(ws + 32768);                // 8192 f32
    unsigned short* reps = (unsigned short*)(ws + 65536);       // 16 MB bf16

    cvt_kernel  <<<(NROWS * DIM / 4) / 256, 256, 0, stream>>>(zi, zj, reps);
    diag_kernel <<<NROWS / 4,              128, 0, stream>>>(zi, zj, diag);
    infonce_main<<<NROWS / 16,             128, 0, stream>>>(reps, diag, nll);
    reduce_mean <<<1,                      256, 0, stream>>>(nll, (float*)d_out);
}
